// Attention_87290915323981
// MI455X (gfx1250) — matrix operand order repaired and checked by reading, not compiled, not run on hardware
//
#include <hip/hip_runtime.h>

typedef __bf16 bf16;
typedef __attribute__((ext_vector_type(16))) __bf16 v16bf;
typedef __attribute__((ext_vector_type(8)))  __bf16 v8bf;
typedef __attribute__((ext_vector_type(8)))  float  v8f;
typedef __attribute__((ext_vector_type(4)))  float  v4f;

#define B_DIM 8
#define N_DIM 4096
#define M_DIM 4096
#define D_DIM 64
#define QROWS 128      // query rows per block (8 waves x 16 rows)
#define KT    64       // keys per tile
#define STR   72       // LDS row stride (bf16 elements), padded vs 64 to dodge bank conflicts
#define NEG_SCORE (-3.0e38f)
// fold 1/temperature and log2(e) into Q so softmax runs in exp2 domain:
// exp((s - m)/T) == exp2(s*QSCALE - m') with QSCALE = log2(e)/T
#define QSCALE (0.125f * 1.44269504088896340736f)

// ---- LDS fragment readers (layouts per CDNA5 ISA 7.12.2, wave32) ----

// B-layout fragment (32x16, 16-bit): lane holds col N=lane%16, elements e hold
// K = e + 16*half (contiguous order chosen here, used for A and B alike) -> 16 contiguous bf16 per lane.
__device__ __forceinline__ v16bf lds_read_contig16(const bf16* p) {
  const v8bf* q = (const v8bf*)p;
  v8bf lo = q[0];
  v8bf hi = q[1];
  v16bf r;
#pragma unroll
  for (int i = 0; i < 8; ++i) { r[i] = lo[i]; r[i + 8] = hi[i]; }
  return r;
}

// A-layout fragment (16x32, 16-bit): lane holds row M=lane%16; element runs
// K = d0..d0+7 and K = d0+16..d0+23 where caller passes d0 = seg + 8*half.
__device__ __forceinline__ v16bf lds_read_afrag(const bf16* rowp, int d0) {
  const v8bf* a = (const v8bf*)(rowp + d0);
  const v8bf* b = (const v8bf*)(rowp + d0 + 16);
  v8bf lo = a[0];
  v8bf hi = b[0];
  v16bf r;
#pragma unroll
  for (int i = 0; i < 8; ++i) { r[i] = lo[i]; r[i + 8] = hi[i]; }
  return r;
}

// pack 8 floats -> one 16B LDS store (ds_store_b128)
__device__ __forceinline__ void lds_store8(bf16* dst, v4f a, v4f b, float scl) {
  v8bf t;
#pragma unroll
  for (int i = 0; i < 4; ++i) {
    t[i]     = (bf16)(a[i] * scl);
    t[i + 4] = (bf16)(b[i] * scl);
  }
  *(v8bf*)dst = t;
}

// Row reductions across the 16 lanes of a half-wave (xor masks keep bit4).
__device__ __forceinline__ float redmax16(float v) {
  v = fmaxf(v, __shfl_xor(v, 1, 32));
  v = fmaxf(v, __shfl_xor(v, 2, 32));
  v = fmaxf(v, __shfl_xor(v, 4, 32));
  v = fmaxf(v, __shfl_xor(v, 8, 32));
  return v;
}
__device__ __forceinline__ float redsum16(float v) {
  v += __shfl_xor(v, 1, 32);
  v += __shfl_xor(v, 2, 32);
  v += __shfl_xor(v, 4, 32);
  v += __shfl_xor(v, 8, 32);
  return v;
}

__global__ __launch_bounds__(256) void fa_wmma_kernel(
    const float* __restrict__ Q, const float* __restrict__ K,
    const float* __restrict__ V, const int* __restrict__ MQ,
    const int* __restrict__ MK, float* __restrict__ O) {
  __shared__ bf16 qLDS[QROWS * STR];      // (Q * QSCALE * mask_q) tile, [qrow][d]
  __shared__ bf16 kLDS[KT * STR];         // K tile, row-major [key][d]
  __shared__ bf16 vLDS[D_DIM * STR];      // V tile, transposed [d][key]
  __shared__ bf16 pLDS[8 * 16 * STR];     // per-wave P bounce [wave][qrow][key]

  const int tid    = threadIdx.x;
  const int wave   = tid >> 5;
  const int lane   = tid & 31;
  const int lane16 = lane & 15;
  const int half   = lane >> 4;
  const int bidx   = blockIdx.y;
  const int q0     = blockIdx.x * QROWS;

  // ---- cooperative load of Q tile; fold softmax scale AND query mask in ----
  {
    const size_t qbase = ((size_t)bidx * N_DIM + q0) * D_DIM;
#pragma unroll
    for (int u = 0; u < 2; ++u) {
      int unit = tid + u * 256;           // 512 units = 128 rows x 4 chunks
      int r = unit >> 2;
      int c = (unit & 3) * 16;
      float scl = QSCALE * (float)MQ[bidx * N_DIM + q0 + r];  // 0 or QSCALE
      const v4f* src = (const v4f*)(Q + qbase + (size_t)r * D_DIM + c);
      v4f x0 = src[0], x1 = src[1], x2 = src[2], x3 = src[3];
      lds_store8(&qLDS[r * STR + c],     x0, x1, scl);
      lds_store8(&qLDS[r * STR + c + 8], x2, x3, scl);
    }
  }
  __syncthreads();

  // ---- per-wave Q fragments (A layout), rows q0 + wave*16 + 0..15 ----
  const bf16* qrow = &qLDS[(wave * 16 + lane16) * STR];
  const v16bf qf0 = lds_read_contig16(qrow + 16 * half);        // d = 0..31
  const v16bf qf1 = lds_read_contig16(qrow + 32 + 16 * half);   // d = 32..63

  v8f zero;
#pragma unroll
  for (int i = 0; i < 8; ++i) zero[i] = 0.f;

  float mrow[8], lrow[8];
  v8f o[4];
#pragma unroll
  for (int r = 0; r < 8; ++r) { mrow[r] = -1.0e30f; lrow[r] = 0.f; }
#pragma unroll
  for (int nt = 0; nt < 4; ++nt) o[nt] = zero;

  bf16* pScr = &pLDS[wave * 16 * STR];

  for (int kb = 0; kb < M_DIM; kb += KT) {
    __syncthreads();  // previous tile's consumers done before overwrite
    // ---- cooperative load of K (row-major) and V (transposed) tiles ----
    {
      int r = tid >> 2;                   // key row 0..63
      int c = (tid & 3) * 16;             // d chunk
      const size_t gbase = ((size_t)bidx * M_DIM + kb + r) * D_DIM + c;
      const v4f* ksrc = (const v4f*)(K + gbase);
      const v4f* vsrc = (const v4f*)(V + gbase);
      v4f k0 = ksrc[0], k1 = ksrc[1], k2 = ksrc[2], k3 = ksrc[3];
      lds_store8(&kLDS[r * STR + c],     k0, k1, 1.f);
      lds_store8(&kLDS[r * STR + c + 8], k2, k3, 1.f);
#pragma unroll
      for (int vv = 0; vv < 4; ++vv) {
        v4f vx = vsrc[vv];
#pragma unroll
        for (int i = 0; i < 4; ++i)
          vLDS[(c + vv * 4 + i) * STR + r] = (bf16)vx[i];
      }
      if (kb + KT < M_DIM) {
        __builtin_prefetch(K + gbase + (size_t)KT * D_DIM, 0, 1);
        __builtin_prefetch(V + gbase + (size_t)KT * D_DIM, 0, 1);
      }
    }
    __syncthreads();

    // ---- S = (Q*QSCALE*mq) * K^T : 4 col tiles of 16 keys, 2 WMMAs each ----
    v8f s[4];
#pragma unroll
    for (int jt = 0; jt < 4; ++jt) {
      const bf16* krow = &kLDS[(jt * 16 + lane16) * STR + 16 * half];
      v16bf kb0 = lds_read_contig16(krow);        // d 0..31
      v16bf kb1 = lds_read_contig16(krow + 32);   // d 32..63
      v8f acc = zero;
      acc = __builtin_amdgcn_wmma_f32_16x16x32_bf16(false, qf0, false, kb0,
                                                    (short)0, acc, false, false);
      acc = __builtin_amdgcn_wmma_f32_16x16x32_bf16(false, qf1, false, kb1,
                                                    (short)0, acc, false, false);
      s[jt] = acc;
    }

    // key-mask flags for this lane's 4 score columns
    int kmask[4];
#pragma unroll
    for (int jt = 0; jt < 4; ++jt)
      kmask[jt] = MK[bidx * M_DIM + kb + jt * 16 + lane16];

    // ---- online softmax update in exp2 domain (row = r + 8*half) ----
    float alpha[8];
    float p[4][8];
#pragma unroll
    for (int r = 0; r < 8; ++r) {
      float t[4];
#pragma unroll
      for (int jt = 0; jt < 4; ++jt)
        t[jt] = (kmask[jt] != 0) ? s[jt][r] : NEG_SCORE;
      float rm = fmaxf(fmaxf(t[0], t[1]), fmaxf(t[2], t[3]));
      rm = redmax16(rm);
      float mnew = fmaxf(mrow[r], rm);
      float al = exp2f(mrow[r] - mnew);        // native v_exp_f32
      mrow[r] = mnew;
      alpha[r] = al;
      float rs = 0.f;
#pragma unroll
      for (int jt = 0; jt < 4; ++jt) {
        float pe = exp2f(t[jt] - mnew);
        p[jt][r] = pe;
        rs += pe;
      }
      lrow[r] = lrow[r] * al + redsum16(rs);
    }

    // ---- bounce P through LDS (C layout f32 -> A layout bf16) ----
#pragma unroll
    for (int jt = 0; jt < 4; ++jt)
#pragma unroll
      for (int r = 0; r < 8; ++r)
        pScr[(r + 8 * half) * STR + jt * 16 + lane16] = (bf16)p[jt][r];

    asm volatile("s_wait_dscnt 0" ::: "memory");  // per-wave DS RAW fence

    // ---- O = diag(alpha)*O + P*V ----
    const bf16* prow = &pScr[lane16 * STR];
    v16bf pf0 = lds_read_contig16(prow + 16 * half);        // keys 0..31
    v16bf pf1 = lds_read_contig16(prow + 32 + 16 * half);   // keys 32..63
#pragma unroll
    for (int nt = 0; nt < 4; ++nt) {
#pragma unroll
      for (int r = 0; r < 8; ++r) o[nt][r] *= alpha[r];
      const bf16* vrow = &vLDS[(nt * 16 + lane16) * STR + 16 * half];
      v16bf vb0 = lds_read_contig16(vrow);        // keys 0..31
      v16bf vb1 = lds_read_contig16(vrow + 32);   // keys 32..63
      o[nt] = __builtin_amdgcn_wmma_f32_16x16x32_bf16(false, pf0, false, vb0,
                                                      (short)0, o[nt], false, false);
      o[nt] = __builtin_amdgcn_wmma_f32_16x16x32_bf16(false, pf1, false, vb1,
                                                      (short)0, o[nt], false, false);
    }
  }

  // ---- epilogue: O /= l, store fp32 ----
  float inv[8];
#pragma unroll
  for (int r = 0; r < 8; ++r) inv[r] = (lrow[r] > 0.f) ? 1.f / lrow[r] : 0.f;
#pragma unroll
  for (int nt = 0; nt < 4; ++nt) {
#pragma unroll
    for (int r = 0; r < 8; ++r) {
      int row = q0 + wave * 16 + r + 8 * half;
      O[((size_t)bidx * N_DIM + row) * D_DIM + nt * 16 + lane16] =
          o[nt][r] * inv[r];
    }
  }
}

extern "C" void kernel_launch(void* const* d_in, const int* in_sizes, int n_in,
                              void* d_out, int out_size, void* d_ws, size_t ws_size,
                              hipStream_t stream) {
  (void)in_sizes; (void)n_in; (void)d_ws; (void)ws_size; (void)out_size;
  const float* Q  = (const float*)d_in[0];
  const float* K  = (const float*)d_in[1];
  const float* V  = (const float*)d_in[2];
  const int*   MQ = (const int*)d_in[3];
  const int*   MK = (const int*)d_in[4];
  float* O = (float*)d_out;

  dim3 grid(N_DIM / QROWS, B_DIM);   // 32 x 8 = 256 workgroups
  dim3 block(256);                   // 8 wave32 waves
  fa_wmma_kernel<<<grid, block, 0, stream>>>(Q, K, V, MQ, MK, O);
}
